// SWABlock_13520557047932
// MI455X (gfx1250) — matrix operand order repaired and checked by reading, not compiled, not run on hardware
//
#include <hip/hip_runtime.h>
#include <hip/hip_bf16.h>

typedef unsigned short ushort_t;
typedef __attribute__((ext_vector_type(16))) __bf16 v16bf;
typedef __attribute__((ext_vector_type(8)))  float  v8f;

#define WMMA_BF16(a, b, c) \
    __builtin_amdgcn_wmma_f32_16x16x32_bf16(false, (a), false, (b), (short)0, (c), false, false)

#define XS_STRIDE 520   // 512 + 8 pad (bf16 elems) -> rows shift 4 banks
#define SM_STRIDE 40    // 32 + 8 pad

__device__ __forceinline__ ushort_t f2bf(float f) {
    unsigned int u = __builtin_bit_cast(unsigned int, f);
    u += 0x7FFFu + ((u >> 16) & 1u);          // round-to-nearest-even
    return (ushort_t)(u >> 16);
}

// Pack two f32 into packed bf16x2 (one v_cvt_pk_bf16_f32 when available)
__device__ __forceinline__ unsigned pack_bf16(float a, float b) {
#if __has_builtin(__builtin_amdgcn_cvt_pk_bf16_f32)
    typedef __attribute__((ext_vector_type(2))) __bf16 v2bf;
    v2bf r = __builtin_amdgcn_cvt_pk_bf16_f32(a, b);
    return __builtin_bit_cast(unsigned, r);
#else
    return (unsigned)f2bf(a) | ((unsigned)f2bf(b) << 16);
#endif
}

// ---- DPP16 butterfly reductions over each 16-lane half-wave (no LDS) ----
// float-typed update_dpp keeps the canonical-float chain (no fcanonicalize maxes)
template <int CTRL>
__device__ __forceinline__ float dpp_f(float v) {
    return __builtin_amdgcn_update_dpp(0.0f, v, CTRL, 0xF, 0xF, true);
}
__device__ __forceinline__ float red_max16(float v) {
    v = fmaxf(v, dpp_f<0xB1>(v));   // quad_perm(1,0,3,2)  : ^1
    v = fmaxf(v, dpp_f<0x4E>(v));   // quad_perm(2,3,0,1)  : ^2
    v = fmaxf(v, dpp_f<0x141>(v));  // row_half_mirror     : ^7 within 8
    v = fmaxf(v, dpp_f<0x140>(v));  // row_mirror          : ^15 within 16
    return v;
}
__device__ __forceinline__ float red_sum16(float v) {
    v += dpp_f<0xB1>(v);
    v += dpp_f<0x4E>(v);
    v += dpp_f<0x141>(v);
    v += dpp_f<0x140>(v);
    return v;
}

union FragBF { uint4 q[2]; v16bf v; };

// A-fragment (16xK tile, K-chunk of 32) from row-major bf16, per CDNA5 layout:
// lanes 0-15: K = kb+0..7, kb+16..23 ; lanes 16-31: K = kb+8..15, kb+24..31
__device__ __forceinline__ v16bf load_a_frag(const ushort_t* base, int stride,
                                             int row0, int kb, int lane) {
    int r  = row0 + (lane & 15);
    int hi = (lane >> 4) << 3;                 // 0 or 8
    const ushort_t* p = base + r * stride + kb + hi;
    FragBF f;
    f.q[0] = *(const uint4*)(p);
    f.q[1] = *(const uint4*)(p + 16);
    return f.v;
}

// B-fragment (Kx16 tile) from row-major "B^T" (row n holds column n of B):
// lanes 0-15: K = kb+0..7, kb+16..23 ; lanes 16-31: K = kb+8..15, kb+24..31
__device__ __forceinline__ v16bf load_b_frag(const ushort_t* base, int stride,
                                             int col0, int kb, int lane) {
    int n  = col0 + (lane & 15);
    int ko = (lane >> 4) << 3;                 // 0 or 8
    const ushort_t* p = base + n * stride + kb + ko;
    FragBF f;
    f.q[0] = *(const uint4*)(p);
    f.q[1] = *(const uint4*)(p + 16);
    return f.v;
}

// Store C/D-layout f32 tile as bf16 into row-major LDS.
// C layout: VGPR i -> (M = i [+8 for lanes 16-31], N = lane & 15)
__device__ __forceinline__ void store_tile_bf16(ushort_t* base, int stride,
                                                int M0, int N0, const v8f acc, int lane) {
    int n = lane & 15;
    int m = M0 + ((lane >> 4) << 3);
    ushort_t* p = base + m * stride + N0 + n;
#pragma unroll
    for (int i = 0; i < 8; ++i) p[i * stride] = f2bf(acc[i]);
}

// Store C tile TRANSPOSED (used for v^T): per-lane rows become contiguous cols -> one b128 store
__device__ __forceinline__ void store_tile_vT(ushort_t* base, int stride,
                                              int d0, int t0, const v8f acc, int lane) {
    int n  = lane & 15;
    int tt = t0 + ((lane >> 4) << 3);
    uint4 u;
    u.x = pack_bf16(acc[0], acc[1]);
    u.y = pack_bf16(acc[2], acc[3]);
    u.z = pack_bf16(acc[4], acc[5]);
    u.w = pack_bf16(acc[6], acc[7]);
    *(uint4*)(base + (d0 + n) * stride + tt) = u;
}

// Row softmax over a 16-row strip held as (c0 = cols 0-15, c1 = cols 16-31).
// Each 16-lane half-wave owns independent rows; DPP16 ops stay inside it.
__device__ __forceinline__ void softmax_rows(v8f& c0, v8f& c1, float scale) {
#pragma unroll
    for (int i = 0; i < 8; ++i) {
        float v0 = c0[i] * scale, v1 = c1[i] * scale;
        float mx = red_max16(fmaxf(v0, v1));
        float e0 = __expf(v0 - mx), e1 = __expf(v1 - mx);
        float inv = __builtin_amdgcn_rcpf(red_sum16(e0 + e1));
        c0[i] = e0 * inv; c1[i] = e1 * inv;
    }
}

// Second softmax: inputs bounded (P in [0,1] + small bias) -> shift-invariant, skip max
__device__ __forceinline__ void softmax_rows_nomax(v8f& c0, v8f& c1) {
#pragma unroll
    for (int i = 0; i < 8; ++i) {
        float e0 = __expf(c0[i]), e1 = __expf(c1[i]);
        float inv = __builtin_amdgcn_rcpf(red_sum16(e0 + e1));
        c0[i] = e0 * inv; c1[i] = e1 * inv;
    }
}

__device__ __forceinline__ void add_bias(v8f& c0, v8f& c1, const float* rpb_s,
                                         int h, int m0, int lane) {
    int col0 = lane & 15, col1 = col0 + 16;
    int rbase = m0 + ((lane >> 4) << 3);
#pragma unroll
    for (int i = 0; i < 8; ++i) {
        int row = rbase + i;
        c0[i] += rpb_s[(row - col0 + 31) * 16 + h];
        c1[i] += rpb_s[(row - col1 + 31) * 16 + h];
    }
}

// ---- kernel 0: one-shot fp32 -> bf16 weight conversion into workspace ----
__global__ void convert_w(const float* __restrict__ qw, const float* __restrict__ ow,
                          ushort_t* __restrict__ dst) {
    int i = (blockIdx.x * 256 + threadIdx.x) * 4;   // 1,048,576 elements total
    float4 v;
    if (i < 786432) v = *(const float4*)(qw + i);
    else            v = *(const float4*)(ow + (i - 786432));
    uint2 t; t.x = pack_bf16(v.x, v.y); t.y = pack_bf16(v.z, v.w);
    *(uint2*)(dst + i) = t;
}

// ---- fused SWA block kernel: 2 windows per workgroup, 8 wave32 ----
__launch_bounds__(256, 1)
__global__ void swa_fused(const float* __restrict__ x,
                          const ushort_t* __restrict__ qw,   // [1536][512] bf16
                          const ushort_t* __restrict__ ow,   // [512][512]  bf16
                          const float* __restrict__ out_b,   // [512]
                          const float* __restrict__ rpb,     // [63][16]
                          float* __restrict__ out) {
    __shared__ ushort_t xs[64 * XS_STRIDE];        // x (2 windows) in bf16
    __shared__ ushort_t attnout[64 * XS_STRIDE];   // attention output in bf16
    __shared__ ushort_t qbuf[8][32 * SM_STRIDE];   // [g*4+hh] q  (row-major tok x d)
    __shared__ ushort_t kbuf[8][32 * SM_STRIDE];   //          k  (row-major tok x d)
    __shared__ ushort_t vtbuf[8][32 * SM_STRIDE];  //          v^T (d x tok)
    __shared__ ushort_t pbuf[8][32 * SM_STRIDE];   // per-wave softmax(P) staging
    __shared__ float    rpb_s[63 * 16];

    const int tid  = threadIdx.x;
    const int lane = tid & 31;
    const int wave = tid >> 5;
    const long long rowBase = (long long)blockIdx.x * 64;   // global token row
    const float SCALE = 0.17677669529663687f;               // 32^-0.5
    const v8f z8 = {0.f,0.f,0.f,0.f,0.f,0.f,0.f,0.f};

    // Phase 0: stage x (fp32 -> bf16) and rpb table
    {
        const float4* xsrc = (const float4*)(x + rowBase * 512);
#pragma unroll 4
        for (int it = 0; it < 32; ++it) {
            int idx = tid + 256 * it;         // 8192 float4 = 64 rows x 128
            int r = idx >> 7, c4 = idx & 127;
            float4 v = xsrc[idx];
            uint2 t; t.x = pack_bf16(v.x, v.y); t.y = pack_bf16(v.z, v.w);
            *(uint2*)(xs + r * XS_STRIDE + c4 * 4) = t;
        }
        for (int i = tid; i < 1008; i += 256) rpb_s[i] = rpb[i];
    }
    __syncthreads();

    // Head batches: 4 heads x 2 windows = 8 attention problems = 1 per wave
#pragma unroll 1
    for (int hb = 0; hb < 4; ++hb) {
        // --- GEMM1: each wave owns the (hh, dh) slice of q, k AND v ---
        // 12 output tiles per wave: load A once per K-chunk, 12 WMMAs against it.
        {
            int hh = wave >> 1;               // head within batch
            int dh = wave & 1;                // d half (0-15 / 16-31)
            const ushort_t* wbase = qw + (long long)(hb * 128 + hh * 32 + dh * 16) * 512;

            v8f acc[3][4];
#pragma unroll
            for (int j = 0; j < 3; ++j)
#pragma unroll
                for (int s = 0; s < 4; ++s) acc[j][s] = z8;

#pragma unroll 2
            for (int kc = 0; kc < 512; kc += 32) {
                v16bf a[4];
#pragma unroll
                for (int s = 0; s < 4; ++s) a[s] = load_a_frag(xs, XS_STRIDE, s * 16, kc, lane);
#pragma unroll
                for (int j = 0; j < 3; ++j) {   // j: 0=q 1=k 2=v sections (stride 512 rows)
                    v16bf b = load_b_frag(wbase + (long long)j * 512 * 512, 512, 0, kc, lane);
#pragma unroll
                    for (int s = 0; s < 4; ++s) acc[j][s] = WMMA_BF16(a[s], b, acc[j][s]);
                }
            }
#pragma unroll
            for (int j = 0; j < 3; ++j)
#pragma unroll
                for (int s = 0; s < 4; ++s) {
                    int g    = s >> 1;        // window within block
                    int tok0 = (s & 1) * 16;
                    int gi   = g * 4 + hh;
                    if (j == 0)      store_tile_bf16(qbuf[gi], SM_STRIDE, tok0, dh * 16, acc[j][s], lane);
                    else if (j == 1) store_tile_bf16(kbuf[gi], SM_STRIDE, tok0, dh * 16, acc[j][s], lane);
                    else             store_tile_vT (vtbuf[gi], SM_STRIDE, dh * 16, tok0, acc[j][s], lane);
                }
        }
        __syncthreads();

        // --- attention: one (window, head) per wave ---
        {
            int gi = wave;
            int hh = wave & 3;
            int h  = hb * 4 + hh;

            v16bf a0 = load_a_frag(qbuf[gi], SM_STRIDE, 0,  0, lane);
            v16bf a1 = load_a_frag(qbuf[gi], SM_STRIDE, 16, 0, lane);
            v16bf b0 = load_b_frag(kbuf[gi], SM_STRIDE, 0,  0, lane);
            v16bf b1 = load_b_frag(kbuf[gi], SM_STRIDE, 16, 0, lane);
            v8f s00 = z8, s01 = z8, s10 = z8, s11 = z8;
            s00 = WMMA_BF16(a0, b0, s00);  s01 = WMMA_BF16(a0, b1, s01);
            s10 = WMMA_BF16(a1, b0, s10);  s11 = WMMA_BF16(a1, b1, s11);

            softmax_rows(s00, s01, SCALE);               // softmax(QK^T * scale)
            softmax_rows(s10, s11, SCALE);
            add_bias(s00, s01, rpb_s, h, 0,  lane);      // + relative position bias
            add_bias(s10, s11, rpb_s, h, 16, lane);
            softmax_rows_nomax(s00, s01);                // second softmax (faithful)
            softmax_rows_nomax(s10, s11);

            store_tile_bf16(pbuf[wave], SM_STRIDE, 0,  0,  s00, lane);
            store_tile_bf16(pbuf[wave], SM_STRIDE, 0,  16, s01, lane);
            store_tile_bf16(pbuf[wave], SM_STRIDE, 16, 0,  s10, lane);
            store_tile_bf16(pbuf[wave], SM_STRIDE, 16, 16, s11, lane);

            v16bf pa0 = load_a_frag(pbuf[wave], SM_STRIDE, 0,  0, lane);
            v16bf pa1 = load_a_frag(pbuf[wave], SM_STRIDE, 16, 0, lane);
            v16bf vb0 = load_b_frag(vtbuf[gi],  SM_STRIDE, 0,  0, lane);
            v16bf vb1 = load_b_frag(vtbuf[gi],  SM_STRIDE, 16, 0, lane);
            v8f o00 = z8, o01 = z8, o10 = z8, o11 = z8;
            o00 = WMMA_BF16(pa0, vb0, o00);  o01 = WMMA_BF16(pa0, vb1, o01);
            o10 = WMMA_BF16(pa1, vb0, o10);  o11 = WMMA_BF16(pa1, vb1, o11);

            int r0 = (wave >> 2) * 32;       // window row base
            int c0 = h * 32;                 // head column base
            store_tile_bf16(attnout, XS_STRIDE, r0,      c0,      o00, lane);
            store_tile_bf16(attnout, XS_STRIDE, r0,      c0 + 16, o01, lane);
            store_tile_bf16(attnout, XS_STRIDE, r0 + 16, c0,      o10, lane);
            store_tile_bf16(attnout, XS_STRIDE, r0 + 16, c0 + 16, o11, lane);
        }
        __syncthreads();
    }

    // --- GEMM3: out = attnout @ out_w^T + out_b ---
    // 2 passes; each pass: wave owns col-tiles {t0, t0+8}, A loaded once per K-chunk.
#pragma unroll 1
    for (int jp = 0; jp < 2; ++jp) {
        int t0 = wave + 16 * jp;
        const ushort_t* wb0 = ow + (long long)(t0 * 16) * 512;
        const ushort_t* wb1 = ow + (long long)((t0 + 8) * 16) * 512;
        v8f acc[2][4];
#pragma unroll
        for (int p = 0; p < 2; ++p)
#pragma unroll
            for (int s = 0; s < 4; ++s) acc[p][s] = z8;

#pragma unroll 2
        for (int kc = 0; kc < 512; kc += 32) {
            v16bf a[4];
#pragma unroll
            for (int s = 0; s < 4; ++s) a[s] = load_a_frag(attnout, XS_STRIDE, s * 16, kc, lane);
            v16bf b0 = load_b_frag(wb0, 512, 0, kc, lane);
#pragma unroll
            for (int s = 0; s < 4; ++s) acc[0][s] = WMMA_BF16(a[s], b0, acc[0][s]);
            v16bf b1 = load_b_frag(wb1, 512, 0, kc, lane);
#pragma unroll
            for (int s = 0; s < 4; ++s) acc[1][s] = WMMA_BF16(a[s], b1, acc[1][s]);
        }

        int n = lane & 15;
#pragma unroll
        for (int p = 0; p < 2; ++p) {
            int t = t0 + p * 8;
            float bias = out_b[t * 16 + n];
#pragma unroll
            for (int s = 0; s < 4; ++s) {
                int m0 = s * 16 + ((lane >> 4) << 3);
                float* po = out + (rowBase + m0) * 512 + t * 16 + n;
#pragma unroll
                for (int i = 0; i < 8; ++i) po[(long long)i * 512] = acc[p][s][i] + bias;
            }
        }
    }
}

extern "C" void kernel_launch(void* const* d_in, const int* in_sizes, int n_in,
                              void* d_out, int out_size, void* d_ws, size_t ws_size,
                              hipStream_t stream) {
    const float* x     = (const float*)d_in[0];
    const float* qkv_w = (const float*)d_in[1];
    const float* out_w = (const float*)d_in[2];
    const float* out_b = (const float*)d_in[3];
    const float* rpb   = (const float*)d_in[4];

    ushort_t* wbf = (ushort_t*)d_ws;   // [0,786432): qkv_w bf16 ; [786432,1048576): out_w bf16
    convert_w<<<1024, 256, 0, stream>>>(qkv_w, out_w, wbf);
    swa_fused<<<2048, 256, 0, stream>>>(x, wbf, wbf + 786432, out_b, rpb, (float*)d_out);
}
